// RSNN_singlelayer_88845693485076
// MI455X (gfx1250) — matrix operand order repaired and checked by reading, not compiled, not run on hardware
//
#include <hip/hip_runtime.h>
#include <hip/hip_bf16.h>

typedef _Float16 f16;
typedef __attribute__((ext_vector_type(16))) _Float16 v16h;
typedef __attribute__((ext_vector_type(8)))  _Float16 v8h;
typedef __attribute__((ext_vector_type(4)))  _Float16 v4h;
typedef __attribute__((ext_vector_type(8)))  float    v8f;
typedef __attribute__((ext_vector_type(4)))  unsigned tdm_u32x4;
typedef __attribute__((ext_vector_type(8)))  int      tdm_i32x8;
typedef __attribute__((ext_vector_type(4)))  int      tdm_i32x4;

#define NN      2048   // neurons
#define TT      512    // timesteps
#define BB      64     // batch
#define MT      16     // batch rows per group (== WMMA M)
#define NGROUP  4      // BB / MT
#define WGPG    16     // workgroups per group
#define NWG     64     // total persistent workgroups
#define NCHUNK  128    // NN / WGPG output columns per WG
#define THREADS 256    // 8 waves of 32

// ---------------------------------------------------------------------------
// One-time prep: effT[n][k] = f16( recurrent[k][n] ), diagonal zeroed.
// LDS-tiled transpose so both global read and write are coalesced.
// ---------------------------------------------------------------------------
__global__ void prep_effT_kernel(const float* __restrict__ rec,
                                 f16* __restrict__ effT) {
  __shared__ float tile[32][33];
  const int bx = blockIdx.x * 32;  // n range on read
  const int by = blockIdx.y * 32;  // k range on read
  const int tx = threadIdx.x, ty = threadIdx.y;
  for (int i = ty; i < 32; i += 8) {
    int k = by + i, n = bx + tx;
    float v = rec[(size_t)k * NN + n];
    if (k == n) v = 0.0f;            // zero self-connections
    tile[i][tx] = v;
  }
  __syncthreads();
  for (int i = ty; i < 32; i += 8) {
    int n = bx + i, k = by + tx;
    effT[(size_t)n * NN + k] = (f16)tile[tx][i];
  }
}

__global__ void init_counter_kernel(unsigned* c) {
  if (threadIdx.x == 0 && blockIdx.x == 0) *c = 0u;
}

// ---------------------------------------------------------------------------
// TDM: DMA a [MT x NN] f16 tile (row stride NN elements) from global to LDS.
// D# per CDNA5 ISA 8.3/8.4: group0 = {count/type, lds_addr, global_addr},
// group1 = {data_size=2B, tensor_dim0=NN, tensor_dim1=MT, tile_dim0=NN,
// tile_dim1=MT, tensor_dim0_stride=NN}. Groups 2/3 zero (<=2D tensor).
// ---------------------------------------------------------------------------
__device__ __forceinline__ void tdm_load_spikes(unsigned lds_byte_off,
                                                const f16* gsrc) {
  unsigned long long ga = (unsigned long long)(uintptr_t)gsrc;
  tdm_u32x4 g0;
  g0[0] = 1u;                                               // count=1, user D#
  g0[1] = lds_byte_off;                                     // lds_addr (bytes)
  g0[2] = (unsigned)ga;                                     // global_addr lo
  g0[3] = ((unsigned)(ga >> 32) & 0x01FFFFFFu) | 0x80000000u; // addr hi | type=2
  tdm_i32x8 g1;
  g1[0] = 0x00010000;                                       // data_size=1 (2B)
  g1[1] = (int)(((unsigned)NN & 0xFFFFu) << 16);            // tensor_dim0 lo16
  g1[2] = (int)((((unsigned)NN >> 16) & 0xFFFFu) |
                (((unsigned)MT & 0xFFFFu) << 16));          // dim0 hi | dim1 lo
  g1[3] = (int)(((unsigned)NN & 0xFFFFu) << 16);            // dim1 hi=0 | tile_dim0
  g1[4] = MT;                                               // tile_dim1, tile_dim2=0
  g1[5] = NN;                                               // tensor_dim0_stride lo32
  g1[6] = 0;                                                // stride hi | dim1_stride
  g1[7] = 0;
  tdm_i32x4 gz = {0, 0, 0, 0};
#if __has_include(<hip/amd_detail/amd_gfx1250_TDM.h>)
  tdm_i32x8 gz8 = {0, 0, 0, 0, 0, 0, 0, 0};
  __builtin_amdgcn_tensor_load_to_lds(g0, g1, gz, gz, gz8, 0);
#else
  __builtin_amdgcn_tensor_load_to_lds(g0, g1, gz, gz, 0);
#endif
}

// ---------------------------------------------------------------------------
// Persistent RSNN kernel: 64 WGs, one global barrier per timestep.
// Group g = blockIdx.x/16 owns batch rows [16g,16g+16); WG w=blockIdx.x%16
// owns output neurons [128w,128(w+1)). mem/spk state lives in registers in
// the WMMA 16x16 f32 C/D layout (8 elems/thread). Spikes are exchanged as
// pre-swizzled f16 rows in xbuf and DMA'd back into LDS by the TDM.
// ---------------------------------------------------------------------------
__global__ void __launch_bounds__(THREADS, 1)
rsnn_kernel(const float* __restrict__ x, const f16* __restrict__ effT,
            f16* __restrict__ xbuf, float* __restrict__ out,
            unsigned* __restrict__ barrier_cnt) {
  // 16 x 2048 f16 spike matrix, XOR-swizzled in 8-half (16B) blocks:
  // half index = m*2048 + ((k & ~7) ^ (m<<3)) + (k & 7)  -> bank-conflict-free
  __shared__ __align__(32) f16 spk_lds[MT * NN];  // exactly 64 KB

  const int tid  = threadIdx.x;
  const int lane = tid & 31;
  const int wave = tid >> 5;                 // 0..7
  const int grp  = blockIdx.x >> 4;          // 0..3
  const int wgi  = blockIdx.x & 15;          // 0..15
  const int col0 = wgi * NCHUNK + wave * 16; // this wave's global column base
  const int mrow = lane & 15;                // M row (A) / N col (B,C,D)
  const int kh   = lane >> 4;                // lane half select
  const int swz  = mrow << 3;                // LDS xor swizzle

  // zero initial spikes in LDS
  for (int i = tid; i < MT * NN / 4; i += THREADS) {
    v4h z = {};
    *(v4h*)&spk_lds[i * 4] = z;
  }

  // per-thread LIF state in C/D tile layout: element v -> (m = v+8*kh, n = mrow)
  float mem[8], spk[8];
#pragma unroll
  for (int v = 0; v < 8; ++v) { mem[v] = 0.0f; spk[v] = 0.0f; }

  // B fragment: lane holds column n = col0+mrow, halves K = k0 + 16*kh + j
  const f16* bptr = effT + (size_t)(col0 + mrow) * NN + 16 * kh;
  const int jcol = col0 + mrow;

  for (int t = 0; t < TT; ++t) {
    __syncthreads();  // LDS spikes for this step are ready (TDM waited in wave 0)

    // ---- GEMM: acc = spk_{t-1}[16 x 2048] @ effT-chunk[2048 x 16] ----
    v8f acc0 = {}; v8f acc1 = {};
#pragma unroll 4
    for (int k0 = 0; k0 < NN; k0 += 64) {
      {
        const int kb = k0;
        v8h a0 = *(const v8h*)&spk_lds[mrow * NN + ((kb + 16 * kh) ^ swz)];
        v8h a1 = *(const v8h*)&spk_lds[mrow * NN + ((kb + 16 * kh + 8) ^ swz)];
        v16h A = __builtin_shufflevector(a0, a1, 0,1,2,3,4,5,6,7,8,9,10,11,12,13,14,15);
        v16h Bm = *(const v16h*)(bptr + kb);
        acc0 = __builtin_amdgcn_wmma_f32_16x16x32_f16(false, A, false, Bm,
                                                      (short)0, acc0, false, false);
      }
      {
        const int kb = k0 + 32;
        v8h a0 = *(const v8h*)&spk_lds[mrow * NN + ((kb + 16 * kh) ^ swz)];
        v8h a1 = *(const v8h*)&spk_lds[mrow * NN + ((kb + 16 * kh + 8) ^ swz)];
        v16h A = __builtin_shufflevector(a0, a1, 0,1,2,3,4,5,6,7,8,9,10,11,12,13,14,15);
        v16h Bm = *(const v16h*)(bptr + kb);
        acc1 = __builtin_amdgcn_wmma_f32_16x16x32_f16(false, A, false, Bm,
                                                      (short)0, acc1, false, false);
      }
    }

    // ---- LIF update + spike emit (state stays in registers) ----
    f16* xb = xbuf + ((size_t)(t & 1) * BB + (size_t)grp * MT) * NN;
#pragma unroll
    for (int v = 0; v < 8; ++v) {
      const int m = v + 8 * kh;
      const int b = grp * MT + m;
      const size_t idx = ((size_t)b * TT + t) * NN + jcol;
      float cur = x[idx] + 0.1f * (acc0[v] + acc1[v]);
      float mm  = 0.96f * mem[v] * (1.0f - spk[v]) + cur;  // hard reset decay
      float s   = (mm - 0.5f >= 0.0f) ? 1.0f : 0.0f;       // Heaviside
      mem[v] = mm; spk[v] = s;
      out[idx] = s;                                        // kernel output
      // f16 spike at producer-side swizzled column, so the linear TDM copy
      // lands it in LDS already in the bank-conflict-free layout
      xb[(size_t)m * NN + ((jcol & ~7) ^ (m << 3)) + (jcol & 7)] = (f16)s;
    }

    // ---- device-wide step barrier (monotonic counter, no reset races) ----
    __threadfence();          // release spikes to agent scope
    __syncthreads();
    if (tid == 0) {
      __hip_atomic_fetch_add(barrier_cnt, 1u, __ATOMIC_ACQ_REL,
                             __HIP_MEMORY_SCOPE_AGENT);
      const unsigned target = (unsigned)(t + 1) * NWG;
      while (__hip_atomic_load(barrier_cnt, __ATOMIC_ACQUIRE,
                               __HIP_MEMORY_SCOPE_AGENT) < target) {
        __builtin_amdgcn_s_sleep(1);
      }
    }
    __syncthreads();
    __threadfence();          // acquire before re-reading exchanged spikes

    // ---- TDM refill: DMA the group's fresh f16 spike tile into LDS ----
    if (t + 1 < TT) {
      if (wave == 0) {
        tdm_load_spikes((unsigned)(uintptr_t)&spk_lds[0],
                        xbuf + ((size_t)(t & 1) * BB + (size_t)grp * MT) * NN);
        __builtin_amdgcn_s_wait_tensorcnt(0);
      }
      // loop-top __syncthreads() releases the other waves after TDM completes
    }
  }
}

// ---------------------------------------------------------------------------
extern "C" void kernel_launch(void* const* d_in, const int* in_sizes, int n_in,
                              void* d_out, int out_size, void* d_ws, size_t ws_size,
                              hipStream_t stream) {
  (void)in_sizes; (void)n_in; (void)out_size; (void)ws_size;
  const float* x   = (const float*)d_in[0];   // [64,512,2048] f32
  const float* rec = (const float*)d_in[1];   // [2048,2048]  f32
  float* out = (float*)d_out;                 // [64,512,2048] f32 spikes

  unsigned* cnt = (unsigned*)d_ws;                         // barrier counter
  f16* xbuf = (f16*)((char*)d_ws + 256);                   // 2x[64][2048] f16 exchange
  f16* effT = (f16*)((char*)d_ws + 256 + 2 * BB * NN * sizeof(f16)); // 8 MB weights

  init_counter_kernel<<<1, 64, 0, stream>>>(cnt);

  dim3 pgrid(NN / 32, NN / 32), pblock(32, 8);
  prep_effT_kernel<<<pgrid, pblock, 0, stream>>>(rec, effT);

  rsnn_kernel<<<NWG, THREADS, 0, stream>>>(x, effT, xbuf, out, cnt);
}
